// Conv2Seq_1297080123472
// MI455X (gfx1250) — matrix operand order repaired and checked by reading, not compiled, not run on hardware
//
#include <hip/hip_runtime.h>
#include <stdint.h>

typedef __bf16 bf16;
typedef bf16  v16bf  __attribute__((ext_vector_type(16)));
typedef bf16  bf16x8 __attribute__((ext_vector_type(8)));
typedef float v8f    __attribute__((ext_vector_type(8)));

#define B_    64
#define LIN   4096
#define DIN   64
#define H_    256
#define L1_   4093
#define L2_   4086
#define L3_   4071
#define TDEC  60
#define DOUT  128

__device__ __forceinline__ float sigf(float x){ return 1.f/(1.f+__expf(-x)); }
__device__ __forceinline__ bf16x8 ld8(const bf16* p){ return *(const bf16x8*)p; }

// ---------------------------------------------------------------- conv stack
__global__ void conv1_k(const float* __restrict__ x, const float* __restrict__ w,
                        const float* __restrict__ bia, float* __restrict__ out){
  __shared__ float wS[8*64*4];
  for(int i=threadIdx.x;i<8*64*4;i+=blockDim.x) wS[i]=w[i];
  __syncthreads();
  int l = blockIdx.x*blockDim.x + threadIdx.x;
  int bb = blockIdx.y;
  if(l>=L1_) return;
  float acc[8];
  #pragma unroll
  for(int co=0;co<8;co++) acc[co]=bia[co];
  const float* xp = x + ((size_t)bb*LIN + l)*DIN;
  for(int k=0;k<4;k++)
    for(int ci=0;ci<DIN;ci++){
      float xv = xp[(size_t)k*DIN+ci];
      #pragma unroll
      for(int co=0;co<8;co++) acc[co] += xv * wS[(co*64+ci)*4+k];
    }
  float* op = out + ((size_t)bb*L1_ + l)*8;
  #pragma unroll
  for(int co=0;co<8;co++) op[co] = fmaxf(acc[co],0.f);
}

__global__ void conv2_k(const float* __restrict__ in, const float* __restrict__ w,
                        const float* __restrict__ bia, float* __restrict__ out){
  __shared__ float wS[16*8*8];
  for(int i=threadIdx.x;i<16*8*8;i+=blockDim.x) wS[i]=w[i];
  __syncthreads();
  int l = blockIdx.x*blockDim.x + threadIdx.x;
  int bb = blockIdx.y;
  if(l>=L2_) return;
  float acc[16];
  #pragma unroll
  for(int co=0;co<16;co++) acc[co]=bia[co];
  const float* ip = in + ((size_t)bb*L1_ + l)*8;
  for(int k=0;k<8;k++)
    for(int ci=0;ci<8;ci++){
      float v = ip[(size_t)k*8+ci];
      #pragma unroll
      for(int co=0;co<16;co++) acc[co] += v * wS[(co*8+ci)*8+k];
    }
  float* op = out + ((size_t)bb*L2_ + l)*16;
  #pragma unroll
  for(int co=0;co<16;co++) op[co] = fmaxf(acc[co],0.f);
}

__global__ void conv3_k(const float* __restrict__ in, const float* __restrict__ w,
                        const float* __restrict__ bia, float* __restrict__ out){
  __shared__ float wS[32*16*16];
  for(int i=threadIdx.x;i<32*16*16;i+=blockDim.x) wS[i]=w[i];
  __syncthreads();
  int l = blockIdx.x*blockDim.x + threadIdx.x;
  int bb = blockIdx.y;
  if(l>=L3_) return;
  float acc[32];
  #pragma unroll
  for(int co=0;co<32;co++) acc[co]=bia[co];
  const float* ip = in + ((size_t)bb*L2_ + l)*16;
  for(int k=0;k<16;k++)
    for(int ci=0;ci<16;ci++){
      float v = ip[(size_t)k*16+ci];
      #pragma unroll
      for(int co=0;co<32;co++) acc[co] += v * wS[(co*16+ci)*16+k];
    }
  float* op = out + ((size_t)bb*L3_ + l)*32;
  #pragma unroll
  for(int co=0;co<32;co++) op[co] = fmaxf(acc[co],0.f);
}

// ------------------------------------------------------------- prep kernels
__global__ void prep_gatew_k(bf16* __restrict__ dst, const float* __restrict__ wih,
                             const float* __restrict__ whh, int Kin, int Khh, int total){
  int i = blockIdx.x*blockDim.x + threadIdx.x;
  if(i>=total) return;
  int K = Kin + Khh;
  int d = i/(1024*K); int rem = i - d*1024*K;
  int n = rem / K;    int k  = rem - n*K;
  float v = (k<Kin) ? wih[((size_t)d*1024+n)*Kin + k]
                    : whh[((size_t)d*1024+n)*Khh + (k-Kin)];
  dst[i] = (bf16)v;
}
__global__ void prep_bias_k(float* __restrict__ dst, const float* __restrict__ a,
                            const float* __restrict__ b, int n){
  int i = blockIdx.x*blockDim.x + threadIdx.x;
  if(i<n) dst[i] = a[i]+b[i];
}
__global__ void prep_bf16_k(bf16* __restrict__ dst, const float* __restrict__ src, int n){
  int i = blockIdx.x*blockDim.x + threadIdx.x;
  if(i<n) dst[i] = (bf16)src[i];
}
__global__ void zero_u32_k(uint32_t* __restrict__ p, int n){
  int i = blockIdx.x*blockDim.x + threadIdx.x;
  if(i<n) p[i]=0;
}

// ---------------------------------------------------- dec1 attention (per step)
// align[b,l] = (attn_w^T h)·enc[b,l] + h·attn_b  (keys_t never materialized)
__global__ void attn_k(const float* __restrict__ enc, const float* __restrict__ aw,
                       const float* __restrict__ ab, bf16* __restrict__ abuf){
  int bb = blockIdx.x;
  int d  = blockIdx.y;
  bf16* arow = abuf + ((size_t)d*B_ + bb)*288;     // [ctx(32) | h(256)]
  __shared__ float hS[256];
  __shared__ float qp[8][32];
  __shared__ float q[32];
  __shared__ float red[256];
  __shared__ float sc[L3_];
  int tid = threadIdx.x;
  hS[tid] = (float)arow[32+tid];
  __syncthreads();
  { // q[c] = sum_j h[j]*aw[j,c]
    int c = tid & 31, seg = tid >> 5;
    float p = 0.f;
    for(int j=seg*32;j<seg*32+32;j++) p += hS[j]*aw[j*32+c];
    qp[seg][c]=p;
  }
  red[tid] = hS[tid]*ab[tid];
  __syncthreads();
  if(tid<32){ float s=0.f; for(int k=0;k<8;k++) s+=qp[k][tid]; q[tid]=s; }
  for(int s=128;s>0;s>>=1){ if(tid<s) red[tid]+=red[tid+s]; __syncthreads(); }
  float hb = red[0];
  __syncthreads();
  // scores + running max
  float mloc = -3.0e38f;
  const float* ep = enc + (size_t)bb*L3_*32;
  for(int l=tid;l<L3_;l+=256){
    const float* e = ep + (size_t)l*32;
    float s = hb;
    #pragma unroll
    for(int c=0;c<32;c++) s += q[c]*e[c];
    sc[l]=s; mloc=fmaxf(mloc,s);
  }
  red[tid]=mloc; __syncthreads();
  for(int s=128;s>0;s>>=1){ if(tid<s) red[tid]=fmaxf(red[tid],red[tid+s]); __syncthreads(); }
  float mx = red[0]; __syncthreads();
  float sloc=0.f;
  for(int l=tid;l<L3_;l+=256){ float e=__expf(sc[l]-mx); sc[l]=e; sloc+=e; }
  red[tid]=sloc; __syncthreads();
  for(int s=128;s>0;s>>=1){ if(tid<s) red[tid]+=red[tid+s]; __syncthreads(); }
  float inv = 1.f/red[0]; __syncthreads();
  { // ctx[c] = (1/S) * sum_l e[l]*enc[b,l,c]
    int c = tid&31, seg=tid>>5;
    float p=0.f;
    for(int l=seg;l<L3_;l+=8) p += sc[l]*ep[(size_t)l*32+c];
    qp[seg][c]=p;
  }
  __syncthreads();
  if(tid<32){ float s=0.f; for(int k=0;k<8;k++) s+=qp[k][tid]; arow[tid]=(bf16)(s*inv); }
}

// ------------------------------- generic LSTM step: gates WMMA + fused cell update
// C[b, g*256+j] = sum_k A[b,k] * W[g*256+j, k]   (A split across two K regions)
// K / K0 are compile-time: region select folds away, all fragment loads become
// global_load_b128 with immediate offsets from 6 hoisted base pointers.
template<int K, int K0>
__global__ __launch_bounds__(32) void lstm_gates_t(
    const bf16* __restrict__ a0, int a0s,
    const bf16* __restrict__ a1, int a1s,
    const bf16* __restrict__ W, const float* __restrict__ bias,
    float* __restrict__ c, bf16* __restrict__ hOut, int hOutStride,
    bf16* __restrict__ seqOut, int pos, int coloff)
{
  constexpr int NK0 = K0/32, NK = K/32;
  int lane = threadIdx.x & 31;
  int r = lane & 15, half = lane >> 4;
  int m0 = blockIdx.x * 16;
  int j0 = blockIdx.y * 16;
  v8f z8 = {0.f,0.f,0.f,0.f,0.f,0.f,0.f,0.f};
  v8f acc[4] = {z8,z8,z8,z8};
  int row = m0 + r;
  // A layout (16-bit): contiguous 16 K-values per lane-half, k = base+half*16..+15
  const bf16* pA0 = a0 + (size_t)row*a0s + half*16;
  const bf16* pA1 = a1 + (size_t)row*a1s + half*16;
  // B from row-major (N,K) weight: contiguous 16 K-values per lane-half
  const bf16* pW0 = W + (size_t)(0*256 + j0 + r)*K + half*16;
  const bf16* pW1 = W + (size_t)(1*256 + j0 + r)*K + half*16;
  const bf16* pW2 = W + (size_t)(2*256 + j0 + r)*K + half*16;
  const bf16* pW3 = W + (size_t)(3*256 + j0 + r)*K + half*16;
  const bf16* pW[4] = {pW0,pW1,pW2,pW3};
  union FU { v16bf v; bf16x8 h[2]; };
  #pragma unroll
  for(int ks=0; ks<NK; ks++){
    FU af;
    if(ks < NK0){ af.h[0] = ld8(pA0 + ks*32);        af.h[1] = ld8(pA0 + ks*32 + 8); }
    else        { af.h[0] = ld8(pA1 + (ks-NK0)*32);  af.h[1] = ld8(pA1 + (ks-NK0)*32 + 8); }
    #pragma unroll
    for(int g=0; g<4; g++){
      FU bfr; bfr.h[0]=ld8(pW[g] + ks*32); bfr.h[1]=ld8(pW[g] + ks*32 + 8);
      acc[g] = __builtin_amdgcn_wmma_f32_16x16x32_bf16(
          false, af.v, false, bfr.v, (short)0, acc[g], false, false);
    }
  }
  int j = j0 + r;
  float bi = bias[0*256+j], bf_ = bias[1*256+j], bg = bias[2*256+j], bo = bias[3*256+j];
  #pragma unroll
  for(int i=0;i<8;i++){
    int m = m0 + half*8 + i;                  // batch index
    size_t ci = (size_t)m*256 + j;
    float iG = sigf (acc[0][i]+bi);
    float fG = sigf (acc[1][i]+bf_);
    float gG = tanhf(acc[2][i]+bg);
    float oG = sigf (acc[3][i]+bo);
    float cn = fG*c[ci] + iG*gG;
    c[ci] = cn;
    float hv = oG*tanhf(cn);
    hOut[(size_t)m*hOutStride + j] = (bf16)hv;
    seqOut[((size_t)m*TDEC + pos)*512 + coloff + j] = (bf16)hv;
  }
}

// ------------------------------------------------------- output projection WMMA
__global__ __launch_bounds__(32) void proj_k(const bf16* __restrict__ A,
                       const bf16* __restrict__ Wt, const float* __restrict__ ob,
                       float* __restrict__ out){
  int lane = threadIdx.x & 31;
  int r = lane & 15, half = lane >> 4;
  int m0 = blockIdx.x*16, n0 = blockIdx.y*16;
  v8f acc = {0.f,0.f,0.f,0.f,0.f,0.f,0.f,0.f};
  union FU { v16bf v; bf16x8 h[2]; };
  const bf16* pA = A  + (size_t)(m0 + r)*512 + half*16;
  const bf16* pB = Wt + (size_t)(n0 + r)*512 + half*16;
  #pragma unroll
  for(int ks=0; ks<16; ks++){
    FU af; af.h[0] = ld8(pA + ks*32); af.h[1] = ld8(pA + ks*32 + 8);
    FU bfr; bfr.h[0]=ld8(pB + ks*32); bfr.h[1]=ld8(pB + ks*32 + 8);
    acc = __builtin_amdgcn_wmma_f32_16x16x32_bf16(
        false, af.v, false, bfr.v, (short)0, acc, false, false);
  }
  int n = n0 + r; float bn = ob[n];
  #pragma unroll
  for(int i=0;i<8;i++)
    out[(size_t)(m0 + half*8 + i)*DOUT + n] = acc[i] + bn;
}

// ============================================================== host driver
extern "C" void kernel_launch(void* const* d_in, const int* in_sizes, int n_in,
                              void* d_out, int out_size, void* d_ws, size_t ws_size,
                              hipStream_t stream) {
  const float* x        = (const float*)d_in[0];
  const float* conv_w1  = (const float*)d_in[1];
  const float* conv_b1  = (const float*)d_in[2];
  const float* conv_w2  = (const float*)d_in[3];
  const float* conv_b2  = (const float*)d_in[4];
  const float* conv_w3  = (const float*)d_in[5];
  const float* conv_b3  = (const float*)d_in[6];
  const float* attn_w   = (const float*)d_in[7];
  const float* attn_b   = (const float*)d_in[8];
  const float* dec1_wih = (const float*)d_in[9];
  const float* dec1_whh = (const float*)d_in[10];
  const float* dec1_bih = (const float*)d_in[11];
  const float* dec1_bhh = (const float*)d_in[12];
  const float* dec2_wih0= (const float*)d_in[13];
  const float* dec2_whh0= (const float*)d_in[14];
  const float* dec2_bih0= (const float*)d_in[15];
  const float* dec2_bhh0= (const float*)d_in[16];
  const float* dec2_wih1= (const float*)d_in[17];
  const float* dec2_whh1= (const float*)d_in[18];
  const float* dec2_bih1= (const float*)d_in[19];
  const float* dec2_bhh1= (const float*)d_in[20];
  const float* out_w    = (const float*)d_in[21];
  const float* out_b    = (const float*)d_in[22];

  char* base = (char*)d_ws; size_t off = 0;
  auto alloc = [&](size_t bytes)->void*{
    void* p = base + off; off = (off + bytes + 255) & ~size_t(255); return p; };

  float* enc  = (float*)alloc((size_t)B_*L3_*32*4);
  float* out1 = (float*)alloc((size_t)B_*L1_*8*4);
  float* out2 = (float*)alloc((size_t)B_*L2_*16*4);
  bf16*  w1c  = (bf16*) alloc((size_t)2*1024*288*2);
  bf16*  w20  = (bf16*) alloc((size_t)2*1024*768*2);
  bf16*  w21  = (bf16*) alloc((size_t)2*1024*768*2);
  bf16*  owb  = (bf16*) alloc((size_t)128*512*2);
  float* b1c  = (float*)alloc((size_t)2048*4);
  float* b20  = (float*)alloc((size_t)2048*4);
  float* b21  = (float*)alloc((size_t)2048*4);
  bf16*  abufA= (bf16*) alloc((size_t)2*B_*288*2);
  bf16*  abufB= (bf16*) alloc((size_t)2*B_*288*2);
  bf16*  hA   = (bf16*) alloc((size_t)2*B_*256*2);
  bf16*  hB   = (bf16*) alloc((size_t)2*B_*256*2);
  float* cb   = (float*)alloc((size_t)2*B_*256*4);
  bf16*  d2in = (bf16*) alloc((size_t)B_*TDEC*512*2);
  bf16*  l0   = (bf16*) alloc((size_t)B_*TDEC*512*2);
  bf16*  l1   = (bf16*) alloc((size_t)B_*TDEC*512*2);
  (void)ws_size; (void)n_in; (void)in_sizes; (void)out_size;

  // conv stack
  conv1_k<<<dim3((L1_+255)/256, B_), 256, 0, stream>>>(x, conv_w1, conv_b1, out1);
  conv2_k<<<dim3((L2_+255)/256, B_), 256, 0, stream>>>(out1, conv_w2, conv_b2, out2);
  conv3_k<<<dim3((L3_+255)/256, B_), 256, 0, stream>>>(out2, conv_w3, conv_b3, enc);

  // weight prep (bf16, [wih|whh] concat)
  { int t=2*1024*288; prep_gatew_k<<<(t+255)/256,256,0,stream>>>(w1c, dec1_wih, dec1_whh, 32, 256, t); }
  { int t=2*1024*768; prep_gatew_k<<<(t+255)/256,256,0,stream>>>(w20, dec2_wih0, dec2_whh0, 512, 256, t); }
  { int t=2*1024*768; prep_gatew_k<<<(t+255)/256,256,0,stream>>>(w21, dec2_wih1, dec2_whh1, 512, 256, t); }
  { int t=128*512;    prep_bf16_k<<<(t+255)/256,256,0,stream>>>(owb, out_w, t); }
  prep_bias_k<<<(2048+255)/256,256,0,stream>>>(b1c, dec1_bih, dec1_bhh, 2048);
  prep_bias_k<<<(2048+255)/256,256,0,stream>>>(b20, dec2_bih0, dec2_bhh0, 2048);
  prep_bias_k<<<(2048+255)/256,256,0,stream>>>(b21, dec2_bih1, dec2_bhh1, 2048);

  // ---- dec1: attention LSTM, 60 steps, 2 directions
  { int n=(2*B_*288*2)/4; zero_u32_k<<<(n+255)/256,256,0,stream>>>((uint32_t*)abufA, n); }
  { int n= 2*B_*256;      zero_u32_k<<<(n+255)/256,256,0,stream>>>((uint32_t*)cb, n); }
  bf16 *aIn = abufA, *aOut = abufB;
  for(int t=0;t<TDEC;t++){
    attn_k<<<dim3(B_,2), 256, 0, stream>>>(enc, attn_w, attn_b, aIn);
    for(int d=0; d<2; d++){
      int pos = d ? (TDEC-1-t) : t;
      lstm_gates_t<288,288><<<dim3(4,16), 32, 0, stream>>>(
          aIn + (size_t)d*B_*288, 288,
          aIn + (size_t)d*B_*288, 288,
          w1c + (size_t)d*1024*288, b1c + d*1024,
          cb  + (size_t)d*B_*256,
          aOut + (size_t)d*B_*288 + 32, 288,
          d2in, pos, d?256:0);
    }
    bf16* tmp=aIn; aIn=aOut; aOut=tmp;
  }

  // ---- dec2 layer0 & layer1: 60 steps, 2 directions, K=768 (512 seq | 256 h)
  bf16* seqIn = d2in; bf16* seqOut = l0;
  const bf16* Ws[2] = { w20, w21 };
  const float* Bs[2] = { b20, b21 };
  for(int layer=0; layer<2; layer++){
    { int n=(2*B_*256*2)/4; zero_u32_k<<<(n+255)/256,256,0,stream>>>((uint32_t*)hA, n); }
    { int n= 2*B_*256;      zero_u32_k<<<(n+255)/256,256,0,stream>>>((uint32_t*)cb, n); }
    bf16 *hIn = hA, *hOut = hB;
    for(int t=0;t<TDEC;t++){
      for(int d=0; d<2; d++){
        int pos = d ? (TDEC-1-t) : t;
        lstm_gates_t<768,512><<<dim3(4,16), 32, 0, stream>>>(
            seqIn + (size_t)pos*512, TDEC*512,
            hIn + (size_t)d*B_*256, 256,
            Ws[layer] + (size_t)d*1024*768, Bs[layer] + d*1024,
            cb + (size_t)d*B_*256,
            hOut + (size_t)d*B_*256, 256,
            seqOut, pos, d?256:0);
      }
      bf16* tmp=hIn; hIn=hOut; hOut=tmp;
    }
    seqIn = l0; seqOut = l1;
  }

  // ---- output projection: (3840,512) @ (128,512)^T + bias
  proj_k<<<dim3(240,8), 32, 0, stream>>>(l1, owb, out_b, (float*)d_out);
}
